// TransformerClassifier_80144089743537
// MI455X (gfx1250) — matrix operand order repaired and checked by reading, not compiled, not run on hardware
//
#include <hip/hip_runtime.h>
#include <hip/hip_bf16.h>

// ---------------------------------------------------------------------------
// TransformerClassifier forward for MI455X (gfx1250, wave32, WMMA).
// Dense math: v_wmma_f32_16x16x32_f16, double-buffered LDS, async B-operand
// copies via global_load_async_to_lds_b128 (ASYNCcnt).
// ---------------------------------------------------------------------------

typedef __attribute__((ext_vector_type(16))) _Float16 v16h;
typedef __attribute__((ext_vector_type(8)))  _Float16 v8h;
typedef __attribute__((ext_vector_type(8)))  float    v8f;
typedef __attribute__((ext_vector_type(4)))  float    v4f;

#define TC_B    4
#define TC_S    2048
#define TC_S1   2049          // with CLS token
#define TC_SP   2112          // padded to multiple of 64
#define TC_MP   (TC_B * TC_SP)
#define TC_D    512
#define TC_H    8
#define TC_DK   64
#define TC_DFF  2048
#define TC_NC   10
#define TC_EPS  1e-5f

#define GEMM_BM 64
#define GEMM_BN 64
#define GEMM_BK 32
#define LDT     40            // LDS row stride in halves (32 + 8 pad, keeps 16B align)

// ---- WMMA fragment loads from LDS ------------------------------------------
// A: 16x32 f16 tile, row-major, ld halves. Lane l<16: row=l, K {0..7,16..23};
// lane>=16: row=l-16, K {8..15,24..31}. (ISA 7.12.2, 16-bit A 16x32)
__device__ __forceinline__ v16h frag_a(const _Float16* base, int ld) {
  const int lane = (int)(threadIdx.x & 31u);
  const int r    = lane & 15;
  const int koff = (lane < 16) ? 0 : 8;
  const _Float16* p = base + r * ld + koff;
  v8h lo = *(const v8h*)(p);
  v8h hi = *(const v8h*)(p + 16);
  v16h f;
#pragma unroll
  for (int i = 0; i < 8; ++i) { f[i] = lo[i]; f[i + 8] = hi[i]; }
  return f;
}
// B operand staged as [N][K] row-major (i.e. B^T). Lane l<16: col N=l holds
// K {0..7,16..23}; lane>=16: N=l-16 holds K {8..15,24..31}.
__device__ __forceinline__ v16h frag_b(const _Float16* base, int ld) {
  const int lane = (int)(threadIdx.x & 31u);
  const int r    = lane & 15;
  const int koff = (lane < 16) ? 0 : 8;
  const _Float16* p = base + r * ld + koff;
  v8h lo = *(const v8h*)(p);
  v8h hi = *(const v8h*)(p + 16);
  v16h f;
#pragma unroll
  for (int i = 0; i < 8; ++i) { f[i] = lo[i]; f[i + 8] = hi[i]; }
  return f;
}

// ---- Generic batched WMMA GEMM: C[M,N] = A[M,K](f32) * B[N,K](f16)^T -------
// 128 threads = 4 waves; each wave owns a 32x32 macro-tile (4 WMMAs/K-step,
// 1:1 fragment-load:WMMA). Double-buffered LDS, one barrier per K-step.
// B tiles are copied global->LDS with the gfx1250 async-to-LDS engine.
// Requirements: M%64==0, N%64==0, K%32==0 (guaranteed by padding).
__global__ void __launch_bounds__(128)
wmma_gemm_kernel(const float* __restrict__ A, const _Float16* __restrict__ B,
                 float* __restrict__ C, const float* __restrict__ bias,
                 int M, int N, int K,
                 long long sA, long long sB, long long sC,
                 float scale, int relu)
{
  __shared__ _Float16 As[2][GEMM_BM * LDT];
  __shared__ _Float16 Bs[2][GEMM_BN * LDT];

  const int bz = blockIdx.z;
  A += (long long)bz * sA;
  B += (long long)bz * sB;
  C += (long long)bz * sC;

  const int m0  = blockIdx.y * GEMM_BM;
  const int n0  = blockIdx.x * GEMM_BN;
  const int tid = (int)threadIdx.x;
  const int w   = tid >> 5;
  const int wm  = (w & 1) * 32;        // wave 32-row band
  const int wn  = (w >> 1) * 32;       // wave 32-col band

  // A staging: 64 rows x 32 f32 -> f16; 16 elements per thread.
  const int ar = tid >> 1;             // 0..63
  const int ac = (tid & 1) * 16;       // 0 or 16
  // B staging: 256 chunks of 8 halves (16B); 2 chunks per thread.
  const int bc0 = tid * 2, bc1 = tid * 2 + 1;
  const int br0 = bc0 >> 2, bh0 = (bc0 & 3) * 8;
  const int br1 = bc1 >> 2, bh1 = (bc1 & 3) * 8;

  v8f acc00 = {}, acc01 = {}, acc10 = {}, acc11 = {};

  auto stage = [&](int kt, int bufIdx) {
    // ---- B tile: async copy global -> LDS (tracked by ASYNCcnt) ----
    {
      const _Float16* g0 = B + (long long)(n0 + br0) * K + (kt + bh0);
      const _Float16* g1 = B + (long long)(n0 + br1) * K + (kt + bh1);
      unsigned int l0 = (unsigned int)(unsigned long long)(&Bs[bufIdx][br0 * LDT + bh0]);
      unsigned int l1 = (unsigned int)(unsigned long long)(&Bs[bufIdx][br1 * LDT + bh1]);
      asm volatile("global_load_async_to_lds_b128 %0, %1, off"
                   :: "v"(l0), "v"((unsigned long long)g0) : "memory");
      asm volatile("global_load_async_to_lds_b128 %0, %1, off"
                   :: "v"(l1), "v"((unsigned long long)g1) : "memory");
    }
    // ---- A tile: global f32 -> regs -> convert -> LDS f16 ----
    {
      const float* ap = A + (long long)(m0 + ar) * K + (kt + ac);
      v4f x0 = *(const v4f*)(ap);
      v4f x1 = *(const v4f*)(ap + 4);
      v4f x2 = *(const v4f*)(ap + 8);
      v4f x3 = *(const v4f*)(ap + 12);
      v8h h0, h1;
#pragma unroll
      for (int i = 0; i < 4; ++i) {
        h0[i]     = (_Float16)x0[i];
        h0[i + 4] = (_Float16)x1[i];
        h1[i]     = (_Float16)x2[i];
        h1[i + 4] = (_Float16)x3[i];
      }
      _Float16* asd = &As[bufIdx][ar * LDT + ac];
      *(v8h*)(asd)     = h0;
      *(v8h*)(asd + 8) = h1;
    }
  };

  // Prologue: fill buffer 0.
  stage(0, 0);
  asm volatile("s_wait_asynccnt 0x0" ::: "memory");
  __syncthreads();

  int buf = 0;
  for (int kt = 0; kt < K; kt += GEMM_BK) {
    const int nkt = kt + GEMM_BK;
    if (nkt < K) stage(nkt, buf ^ 1);   // prefetch next tile into other buffer

    const v16h a0 = frag_a(&As[buf][wm * LDT], LDT);
    const v16h a1 = frag_a(&As[buf][(wm + 16) * LDT], LDT);
    const v16h b0 = frag_b(&Bs[buf][wn * LDT], LDT);
    const v16h b1 = frag_b(&Bs[buf][(wn + 16) * LDT], LDT);
    acc00 = __builtin_amdgcn_wmma_f32_16x16x32_f16(false, a0, false, b0,
                                                   (short)0, acc00, false, false);
    acc01 = __builtin_amdgcn_wmma_f32_16x16x32_f16(false, a0, false, b1,
                                                   (short)0, acc01, false, false);
    acc10 = __builtin_amdgcn_wmma_f32_16x16x32_f16(false, a1, false, b0,
                                                   (short)0, acc10, false, false);
    acc11 = __builtin_amdgcn_wmma_f32_16x16x32_f16(false, a1, false, b1,
                                                   (short)0, acc11, false, false);

    if (nkt < K) {
      asm volatile("s_wait_asynccnt 0x0" ::: "memory");
      __syncthreads();
      buf ^= 1;
    }
  }

  // Store: VGPR i -> row (+i) for lanes 0-15, row (+8+i) for lanes 16-31.
  const int lane  = tid & 31;
  const int rb    = m0 + wm + ((lane < 16) ? 0 : 8);
  const int col   = n0 + wn + (lane & 15);
  const float bi0 = bias ? bias[col]      : 0.0f;
  const float bi1 = bias ? bias[col + 16] : 0.0f;
#pragma unroll
  for (int i = 0; i < 8; ++i) {
    float v00 = acc00[i] * scale + bi0;
    float v01 = acc01[i] * scale + bi1;
    float v10 = acc10[i] * scale + bi0;
    float v11 = acc11[i] * scale + bi1;
    if (relu) {
      v00 = fmaxf(v00, 0.0f); v01 = fmaxf(v01, 0.0f);
      v10 = fmaxf(v10, 0.0f); v11 = fmaxf(v11, 0.0f);
    }
    C[(long long)(rb + i) * N + col]           = v00;
    C[(long long)(rb + i) * N + col + 16]      = v01;
    C[(long long)(rb + 16 + i) * N + col]      = v10;
    C[(long long)(rb + 16 + i) * N + col + 16] = v11;
  }
}

// ---- Embedding + CLS + sinusoidal PE, with zero padding --------------------
__global__ void __launch_bounds__(256)
embed_pe_kernel(const int* __restrict__ tok, const float* __restrict__ emb,
                const float* __restrict__ cls, float* __restrict__ out)
{
  const long long idx = (long long)blockIdx.x * 256 + threadIdx.x;
  const long long total = (long long)TC_MP * TC_D;
  if (idx >= total) return;
  const int d = (int)(idx % TC_D);
  const int s = (int)((idx / TC_D) % TC_SP);
  const int b = (int)(idx / ((long long)TC_D * TC_SP));
  float v = 0.0f;
  if (s < TC_S1) {
    const float freq = __expf(-(float)(d & ~1) * (9.210340371976184f / (float)TC_D));
    const float ang  = (float)s * freq;
    const float pe   = (d & 1) ? __cosf(ang) : __sinf(ang);
    if (s == 0) {
      v = cls[d] + pe;
    } else {
      const int t = tok[b * TC_S + (s - 1)];
      v = emb[(long long)t * TC_D + d] + pe;
    }
  }
  out[idx] = v;
}

// ---- f32 -> f16 (weights) --------------------------------------------------
__global__ void __launch_bounds__(256)
f32_to_f16_kernel(const float* __restrict__ s, _Float16* __restrict__ d, int n)
{
  const int i = blockIdx.x * 256 + threadIdx.x;
  if (i < n) d[i] = (_Float16)s[i];
}

// ---- head split: [B*Sp, D] -> [B,H,Sp,DK], zero the seq padding ------------
__global__ void __launch_bounds__(256)
split_heads_f32_kernel(const float* __restrict__ src, float* __restrict__ dst)
{
  const long long idx = (long long)blockIdx.x * 256 + threadIdx.x;
  const long long total = (long long)TC_MP * TC_D;
  if (idx >= total) return;
  const int d = (int)(idx % TC_DK);
  const int s = (int)((idx / TC_DK) % TC_SP);
  const int h = (int)((idx / ((long long)TC_DK * TC_SP)) % TC_H);
  const int b = (int)(idx / ((long long)TC_DK * TC_SP * TC_H));
  dst[idx] = (s < TC_S1)
           ? src[((long long)(b * TC_SP + s)) * TC_D + h * TC_DK + d] : 0.0f;
}
__global__ void __launch_bounds__(256)
split_heads_f16_kernel(const float* __restrict__ src, _Float16* __restrict__ dst)
{
  const long long idx = (long long)blockIdx.x * 256 + threadIdx.x;
  const long long total = (long long)TC_MP * TC_D;
  if (idx >= total) return;
  const int d = (int)(idx % TC_DK);
  const int s = (int)((idx / TC_DK) % TC_SP);
  const int h = (int)((idx / ((long long)TC_DK * TC_SP)) % TC_H);
  const int b = (int)(idx / ((long long)TC_DK * TC_SP * TC_H));
  dst[idx] = (s < TC_S1)
           ? (_Float16)src[((long long)(b * TC_SP + s)) * TC_D + h * TC_DK + d]
           : (_Float16)0.0f;
}
// V transposed per head: [B,H,DK,Sp] (so attn @ V = P * (V^T)^T fits A*B^T GEMM)
__global__ void __launch_bounds__(256)
transpose_v_f16_kernel(const float* __restrict__ src, _Float16* __restrict__ dst)
{
  const long long idx = (long long)blockIdx.x * 256 + threadIdx.x;
  const long long total = (long long)TC_MP * TC_D;
  if (idx >= total) return;
  const int s = (int)(idx % TC_SP);
  const int d = (int)((idx / TC_SP) % TC_DK);
  const int h = (int)((idx / ((long long)TC_SP * TC_DK)) % TC_H);
  const int b = (int)(idx / ((long long)TC_SP * TC_DK * TC_H));
  dst[idx] = (s < TC_S1)
           ? (_Float16)src[((long long)(b * TC_SP + s)) * TC_D + h * TC_DK + d]
           : (_Float16)0.0f;
}
// merge heads: [B,H,Sp,DK] -> [B*Sp, D]
__global__ void __launch_bounds__(256)
merge_heads_kernel(const float* __restrict__ src, float* __restrict__ dst)
{
  const long long idx = (long long)blockIdx.x * 256 + threadIdx.x;
  const long long total = (long long)TC_MP * TC_D;
  if (idx >= total) return;
  const int d = (int)(idx % TC_D);
  const int s = (int)((idx / TC_D) % TC_SP);
  const int b = (int)(idx / ((long long)TC_D * TC_SP));
  const int h = d / TC_DK, dd = d % TC_DK;
  dst[idx] = src[(((long long)(b * TC_H + h)) * TC_SP + s) * TC_DK + dd];
}

// ---- row softmax with key-padding mask (cols >= valid -> 0) ----------------
__global__ void __launch_bounds__(256)
softmax_kernel(float* __restrict__ S, int cols, int valid)
{
  float* row = S + (long long)blockIdx.x * cols;
  __shared__ float red[256];
  const int tid = (int)threadIdx.x;
  float m = -1e30f;
  for (int c = tid; c < valid; c += 256) m = fmaxf(m, row[c]);
  red[tid] = m; __syncthreads();
  for (int st = 128; st > 0; st >>= 1) {
    if (tid < st) red[tid] = fmaxf(red[tid], red[tid + st]);
    __syncthreads();
  }
  m = red[0]; __syncthreads();
  float sum = 0.0f;
  for (int c = tid; c < valid; c += 256) {
    const float e = __expf(row[c] - m);
    row[c] = e; sum += e;
  }
  red[tid] = sum; __syncthreads();
  for (int st = 128; st > 0; st >>= 1) {
    if (tid < st) red[tid] += red[tid + st];
    __syncthreads();
  }
  const float inv = 1.0f / red[0];
  for (int c = tid; c < cols; c += 256)
    row[c] = (c < valid) ? row[c] * inv : 0.0f;
}

// ---- fused residual add + LayerNorm ----------------------------------------
__global__ void __launch_bounds__(256)
add_ln_kernel(const float* __restrict__ X, const float* __restrict__ Y,
              const float* __restrict__ g, const float* __restrict__ bb,
              float* __restrict__ O, int D)
{
  const long long r = blockIdx.x;
  const float* xr = X + r * D;
  const float* yr = Y + r * D;
  float* orow = O + r * D;
  __shared__ float rs[256];
  __shared__ float rq[256];
  const int tid = (int)threadIdx.x;
  float s = 0.0f, q = 0.0f;
  for (int c = tid; c < D; c += 256) {
    const float t = xr[c] + yr[c];
    orow[c] = t; s += t; q += t * t;
  }
  rs[tid] = s; rq[tid] = q; __syncthreads();
  for (int st = 128; st > 0; st >>= 1) {
    if (tid < st) { rs[tid] += rs[tid + st]; rq[tid] += rq[tid + st]; }
    __syncthreads();
  }
  const float mu  = rs[0] / (float)D;
  const float var = rq[0] / (float)D - mu * mu;
  const float inv = rsqrtf(var + TC_EPS);
  for (int c = tid; c < D; c += 256)
    orow[c] = (orow[c] - mu) * inv * g[c] + bb[c];
}

// ---- classifier head on CLS token ------------------------------------------
__global__ void __launch_bounds__(64)
head_kernel(const float* __restrict__ h, const float* __restrict__ w,
            const float* __restrict__ bias, float* __restrict__ out)
{
  const int c = blockIdx.x, b = blockIdx.y;
  const float* xr = h + (long long)b * TC_SP * TC_D;   // s == 0 row
  const float* wr = w + (long long)c * TC_D;
  __shared__ float red[64];
  const int tid = (int)threadIdx.x;
  float s = 0.0f;
  for (int d = tid; d < TC_D; d += 64) s += xr[d] * wr[d];
  red[tid] = s; __syncthreads();
  for (int st = 32; st > 0; st >>= 1) {
    if (tid < st) red[tid] += red[tid + st];
    __syncthreads();
  }
  if (tid == 0) out[b * TC_NC + c] = red[0] + bias[c];
}

// ---------------------------------------------------------------------------
extern "C" void kernel_launch(void* const* d_in, const int* in_sizes, int n_in,
                              void* d_out, int out_size, void* d_ws, size_t ws_size,
                              hipStream_t stream) {
  (void)in_sizes; (void)n_in; (void)out_size; (void)ws_size;

  // Inputs in setup_inputs() dict order (recursive insertion order).
  const int*   x    = (const int*)  d_in[0];
  const float* emb  = (const float*)d_in[1];
  const float* cls  = (const float*)d_in[2];
  const float* fcw  = (const float*)d_in[3];
  const float* fcb  = (const float*)d_in[4];

  // Workspace carve (256B aligned slices).
  char* wp = (char*)d_ws;
  auto carve = [&](size_t bytes) -> void* {
    void* r = (void*)wp;
    wp += (bytes + 255) & ~(size_t)255;
    return r;
  };
  const size_t actF = (size_t)TC_MP * TC_D * sizeof(float);
  const size_t actH = (size_t)TC_MP * TC_D * sizeof(_Float16);

  float*     hbuf   = (float*)carve(actF);                       // residual stream
  float*     tq     = (float*)carve(actF);
  float*     tk     = (float*)carve(actF);
  float*     tv     = (float*)carve(actF);
  float*     qh     = (float*)carve(actF);                       // [B,H,Sp,DK]
  float*     ctx    = (float*)carve(actF);                       // [B,H,Sp,DK]
  float*     h1     = (float*)carve(actF);
  _Float16*  kh     = (_Float16*)carve(actH);                    // [B,H,Sp,DK]
  _Float16*  vT     = (_Float16*)carve(actH);                    // [B,H,DK,Sp]
  float*     scores = (float*)carve((size_t)TC_H * TC_SP * TC_SP * sizeof(float));
  float*     ffa    = (float*)carve((size_t)TC_MP * TC_DFF * sizeof(float));
  _Float16*  wq16   = (_Float16*)carve((size_t)TC_D * TC_D * 2);
  _Float16*  wk16   = (_Float16*)carve((size_t)TC_D * TC_D * 2);
  _Float16*  wv16   = (_Float16*)carve((size_t)TC_D * TC_D * 2);
  _Float16*  wo16   = (_Float16*)carve((size_t)TC_D * TC_D * 2);
  _Float16*  wf116  = (_Float16*)carve((size_t)TC_DFF * TC_D * 2);
  _Float16*  wf216  = (_Float16*)carve((size_t)TC_D * TC_DFF * 2);
  float*     ctxm   = tq;   // free after head-split
  float*     aout   = tk;
  float*     ffo    = tv;

  const long long elemTotal = (long long)TC_MP * TC_D;
  const int elemBlocks = (int)((elemTotal + 255) / 256);

  auto gemm = [&](const float* A, const _Float16* Bm, float* Cm, const float* bias,
                  int M, int N, int K, long long sA, long long sB, long long sC,
                  int nbatch, float scale, int relu) {
    dim3 grid((unsigned)(N / GEMM_BN), (unsigned)(M / GEMM_BM), (unsigned)nbatch);
    wmma_gemm_kernel<<<grid, 128, 0, stream>>>(A, Bm, Cm, bias, M, N, K,
                                               sA, sB, sC, scale, relu);
  };
  auto conv16 = [&](const float* s, _Float16* d, int n) {
    f32_to_f16_kernel<<<(n + 255) / 256, 256, 0, stream>>>(s, d, n);
  };

  // Embedding + CLS + PE (zero padded rows).
  embed_pe_kernel<<<elemBlocks, 256, 0, stream>>>(x, emb, cls, hbuf);

  for (int l = 0; l < 2; ++l) {
    const int base = 5 + l * 16;
    const float* wqw = (const float*)d_in[base + 0];
    const float* wqb = (const float*)d_in[base + 1];
    const float* wkw = (const float*)d_in[base + 2];
    const float* wkb = (const float*)d_in[base + 3];
    const float* wvw = (const float*)d_in[base + 4];
    const float* wvb = (const float*)d_in[base + 5];
    const float* wow = (const float*)d_in[base + 6];
    const float* wob = (const float*)d_in[base + 7];
    const float* f1w = (const float*)d_in[base + 8];
    const float* f1b = (const float*)d_in[base + 9];
    const float* f2w = (const float*)d_in[base + 10];
    const float* f2b = (const float*)d_in[base + 11];
    const float* g1  = (const float*)d_in[base + 12];
    const float* b1  = (const float*)d_in[base + 13];
    const float* g2  = (const float*)d_in[base + 14];
    const float* b2  = (const float*)d_in[base + 15];

    conv16(wqw, wq16, TC_D * TC_D);
    conv16(wkw, wk16, TC_D * TC_D);
    conv16(wvw, wv16, TC_D * TC_D);
    conv16(wow, wo16, TC_D * TC_D);
    conv16(f1w, wf116, TC_DFF * TC_D);
    conv16(f2w, wf216, TC_D * TC_DFF);

    // Q/K/V projections: [Mp,D] x [D,D]^T
    gemm(hbuf, wq16, tq, wqb, TC_MP, TC_D, TC_D, 0, 0, 0, 1, 1.0f, 0);
    gemm(hbuf, wk16, tk, wkb, TC_MP, TC_D, TC_D, 0, 0, 0, 1, 1.0f, 0);
    gemm(hbuf, wv16, tv, wvb, TC_MP, TC_D, TC_D, 0, 0, 0, 1, 1.0f, 0);

    split_heads_f32_kernel<<<elemBlocks, 256, 0, stream>>>(tq, qh);
    split_heads_f16_kernel<<<elemBlocks, 256, 0, stream>>>(tk, kh);
    transpose_v_f16_kernel<<<elemBlocks, 256, 0, stream>>>(tv, vT);

    // Attention, batched over heads; loop batch dim to bound the S*S buffer.
    for (int b = 0; b < TC_B; ++b) {
      const long long ho = (long long)b * TC_H * TC_SP * TC_DK;
      // scores = (Q K^T) / sqrt(DK) : M=Sp, N=Sp, K=DK per head
      gemm(qh + ho, kh + ho, scores, nullptr,
           TC_SP, TC_SP, TC_DK,
           (long long)TC_SP * TC_DK, (long long)TC_SP * TC_DK,
           (long long)TC_SP * TC_SP, TC_H, 0.125f, 0);
      softmax_kernel<<<TC_H * TC_SP, 256, 0, stream>>>(scores, TC_SP, TC_S1);
      // ctx = P V : M=Sp, N=DK, K=Sp per head (V stored transposed)
      gemm(scores, vT + ho, ctx + ho, nullptr,
           TC_SP, TC_DK, TC_SP,
           (long long)TC_SP * TC_SP, (long long)TC_DK * TC_SP,
           (long long)TC_SP * TC_DK, TC_H, 1.0f, 0);
    }

    merge_heads_kernel<<<elemBlocks, 256, 0, stream>>>(ctx, ctxm);
    gemm(ctxm, wo16, aout, wob, TC_MP, TC_D, TC_D, 0, 0, 0, 1, 1.0f, 0);
    add_ln_kernel<<<TC_MP, 256, 0, stream>>>(hbuf, aout, g1, b1, h1, TC_D);

    // FFN
    gemm(h1, wf116, ffa, f1b, TC_MP, TC_DFF, TC_D, 0, 0, 0, 1, 1.0f, 1);
    gemm(ffa, wf216, ffo, f2b, TC_MP, TC_D, TC_DFF, 0, 0, 0, 1, 1.0f, 0);
    add_ln_kernel<<<TC_MP, 256, 0, stream>>>(h1, ffo, g2, b2, hbuf, TC_D);
  }

  head_kernel<<<dim3(TC_NC, TC_B), 64, 0, stream>>>(hbuf, fcw, fcb, (float*)d_out);
}
